// HGCNEmbeddingContinuous_71133248356948
// MI455X (gfx1250) — matrix operand order repaired and checked by reading, not compiled, not run on hardware
//
#include <hip/hip_runtime.h>
#include <hip/hip_bf16.h>
#include <math.h>

// ---------- problem constants ----------
#define NB 256          // batch
#define PP 64           // P
#define NNB 32          // neighbors per node
#define NPE 4           // nodes per hyperedge
#define DD 64           // dim
#define E2 128          // 2*D
#define H1 96           // 3*D/2
#define MROWS (NB*PP)   // 16384
#define PL (PP*NNB)     // 2048 (P*NNB)

typedef _Float16 f16x16 __attribute__((ext_vector_type(16)));
typedef float    f32x8  __attribute__((ext_vector_type(8)));

// ---- fast transcendental helpers (hardware V_TANH_F32 on gfx1250) ----
__device__ __forceinline__ float fast_tanh(float x) {
#if __has_builtin(__builtin_amdgcn_tanhf)
    return __builtin_amdgcn_tanhf(x);
#elif __has_builtin(__builtin_amdgcn_tanh_f32)
    return __builtin_amdgcn_tanh_f32(x);
#else
    return tanhf(x);
#endif
}

// Load WMMA A fragment from row-major f16 LDS tile.
// Lane l: row = rowBase + (l&15); half hl = l>>4;
//   e<8  -> K = kBase + hl*8 + e
//   e>=8 -> K = kBase + 16 + hl*8 + (e-8)
__device__ __forceinline__ f16x16 load_a_frag(const _Float16* A, int stride,
                                              int rowBase, int kBase, int lane) {
    f16x16 a;
    const int mrow = rowBase + (lane & 15);
    const int hl = lane >> 4;
    const _Float16* ap = A + mrow * stride + kBase + hl * 8;
#pragma unroll
    for (int e = 0; e < 8; ++e) { a[e] = ap[e]; a[e + 8] = ap[e + 16]; }
    return a;
}

// B fragment is pre-staged contiguously: 16 halfs per lane-slot.
__device__ __forceinline__ f16x16 load_b_frag(const _Float16* Bfrag) {
    f16x16 b;
#pragma unroll
    for (int e = 0; e < 16; ++e) b[e] = Bfrag[e];
    return b;
}

__device__ __forceinline__ f32x8 zero8() {
    f32x8 c = {0.f, 0.f, 0.f, 0.f, 0.f, 0.f, 0.f, 0.f};
    return c;
}

// Stage weight matrix W[outDim][inDim] (row-major f32) into WMMA-B fragment
// layout: slot s = (kt*NT + nt)*32 + lane, element e holds
// W[nt*16 + (lane&15)][kt*32 + (lane>>4)*8 + (e&7) + (e>>3)*16]  (B[k][n] = W[n][k], i.e. x@W^T)
__device__ __forceinline__ void stage_w_frag(_Float16* dstBase, const float* W,
                                             int inDim, int NT, int nslots, int tid) {
    for (int s = tid; s < nslots; s += 256) {
        const int kt = s / (NT * 32);
        const int rem = s - kt * (NT * 32);
        const int nt = rem >> 5;
        const int ln = rem & 31;
        const int n = nt * 16 + (ln & 15);
        const int kb = kt * 32 + (ln >> 4) * 8;
        _Float16* dst = dstBase + s * 16;
        const float* src = W + n * inDim + kb;
#pragma unroll
        for (int e = 0; e < 16; ++e) dst[e] = (_Float16)src[(e & 7) + ((e >> 3) << 4)];
    }
}

// =====================================================================
// Kernel 1: per 4 query rows — edge embed (WMMA) + fused attention path.
// Writes attn_out [M,128] f32 into workspace.
// =====================================================================
__global__ void __launch_bounds__(256)
hgcn_attn_kernel(const float* __restrict__ memory,
                 const float* __restrict__ times,
                 const float* __restrict__ basis_freq,
                 const float* __restrict__ phase,
                 const float* __restrict__ Wh, const float* __restrict__ bh,
                 const float* __restrict__ Wq, const float* __restrict__ bq,
                 const float* __restrict__ Wk,
                 const float* __restrict__ Wv, const float* __restrict__ bv,
                 const float* __restrict__ Wo, const float* __restrict__ bo,
                 const int* __restrict__ nodes,
                 const int* __restrict__ h_index,
                 const int* __restrict__ mask,
                 float* __restrict__ attnOut) {
    __shared__ _Float16 sKV[128 * 128];   // kv_in f16: [:,0:64]=edge, [:,64:128]=tenc (32KB)
    __shared__ _Float16 sWf[64 * 64];     // Wh fragments (8KB)
    __shared__ float    sU[4096];         // [0,2048)=qt (aliased as sX f16 early), [2048,4096)=z (16KB)
    __shared__ float    sU2[512];         // q_in -> scores/attn (2KB)
    __shared__ float    sQ[512];          // q -> ctx (2KB)
    __shared__ float    sBF[64], sPH[64];
    __shared__ int      sInv[4];

    const int tid = threadIdx.x;
    const int lane = tid & 31;
    const int wave = tid >> 5;
    const int m0 = blockIdx.x * 4;
    _Float16* sX = (_Float16*)sU;         // 128 x 64 f16 edge-mean inputs (phase 1-2 only)

    // ---- Phase 0: stage Wh fragments (2 ktiles x 4 ntiles), freq/phase ----
    stage_w_frag(sWf, Wh, DD, 4, 2 * 4 * 32, tid);
    if (tid < 64) { sBF[tid] = basis_freq[tid]; sPH[tid] = phase[tid]; }

    // ---- Phase 1: gather + mean of NPE member rows -> sX (f16) ----
    {
        const int e_loc = tid & 127;
        const int ch = tid >> 7;                  // 0/1: which 8 float4-chunks
        const int r = e_loc >> 5, j = e_loc & 31;
        const int m = m0 + r, bb = m >> 6, p = m & 63;
        const int el = p * NNB + j;
        const int* idp = h_index + (size_t)bb * (2 * PL * NPE) + 4 * el;
        const int i0 = idp[0], i1 = idp[1], i2 = idp[2], i3 = idp[3];
        const float4* r0 = (const float4*)(memory + (size_t)i0 * DD);
        const float4* r1 = (const float4*)(memory + (size_t)i1 * DD);
        const float4* r2 = (const float4*)(memory + (size_t)i2 * DD);
        const float4* r3 = (const float4*)(memory + (size_t)i3 * DD);
#pragma unroll
        for (int c = 0; c < 8; ++c) {
            const int cc = ch * 8 + c;
            float4 a = r0[cc], b = r1[cc], g = r2[cc], d = r3[cc];
            _Float16* dst = sX + e_loc * DD + cc * 4;
            dst[0] = (_Float16)(0.25f * (a.x + b.x + g.x + d.x));
            dst[1] = (_Float16)(0.25f * (a.y + b.y + g.y + d.y));
            dst[2] = (_Float16)(0.25f * (a.z + b.z + g.z + d.z));
            dst[3] = (_Float16)(0.25f * (a.w + b.w + g.w + d.w));
        }
    }
    __syncthreads();

    // ---- Phase 2: edge GEMM [128,64]x[64,64] via WMMA; tanh -> sKV[:,0:64] ----
    for (int t = wave; t < 32; t += 8) {          // 8 row-tiles x 4 col-tiles
        const int rt = t >> 2, nt = t & 3;
        f32x8 c = zero8();
#pragma unroll
        for (int kt = 0; kt < 2; ++kt) {
            f16x16 a = load_a_frag(sX, DD, rt * 16, kt * 32, lane);
            f16x16 b = load_b_frag(sWf + ((kt * 4 + nt) * 32 + lane) * 16);
            c = __builtin_amdgcn_wmma_f32_16x16x32_f16(false, a, false, b,
                                                       (short)0, c, false, false);
        }
        const int col = nt * 16 + (lane & 15);
        const float bhv = bh[col];
#pragma unroll
        for (int r = 0; r < 8; ++r) {
            const int row = rt * 16 + (lane >> 4) * 8 + r;
            sKV[row * E2 + col] = (_Float16)fast_tanh(c[r] + bhv);
        }
    }
    // ---- Phase 2b: time encoding -> sKV[:,64:128] ----
    for (int idx = tid; idx < 128 * 64; idx += 256) {
        const int e_loc = idx >> 6, dd = idx & 63;
        const int r = e_loc >> 5, j = e_loc & 31;
        const int m = m0 + r, bb = m >> 6, p = m & 63;
        const float tt = times[bb * PL + p * NNB + j];
        sKV[e_loc * E2 + 64 + dd] = (_Float16)cosf(tt * sBF[dd] + sPH[dd]);
    }
    // ---- Phase 3a: q_in = [memory[nodes[m]], cos(phase)] -> sU2 ----
    {
        const int r = tid >> 6, dd = tid & 63;
        const int m = m0 + r, bb = m >> 6, p = m & 63;
        const int nid = nodes[bb * PP + p];
        sU2[r * E2 + dd] = memory[(size_t)nid * DD + dd];
        sU2[r * E2 + 64 + dd] = cosf(sPH[dd]);
    }
    __syncthreads();

    // ---- Phase 3b: q = Wq @ q_in + bq  (4 rows x 128) ----
    for (int s = tid; s < 512; s += 256) {
        const int r = s >> 7, o = s & 127;
        float acc = bq[o];
        const float* w = Wq + o * E2;
        const float* xin = sU2 + r * E2;
#pragma unroll 8
        for (int i = 0; i < E2; ++i) acc += w[i] * xin[i];
        sQ[s] = acc;
    }
    __syncthreads();

    // ---- Phase 4: qt[r][h][:] = Wk_h^T q_h  (bk drops out of softmax) ----
    for (int s = tid; s < 2048; s += 256) {
        const int r = s >> 9, h = (s >> 7) & 3, jj = s & 127;
        float acc = 0.f;
        const float* q = sQ + r * E2 + h * 32;
#pragma unroll 8
        for (int e = 0; e < 32; ++e) acc += q[e] * Wk[(h * 32 + e) * E2 + jj];
        sU[s] = acc;                               // layout r*512 + h*128 + jj == s
    }
    __syncthreads();

    // ---- Phase 5: scores = (qt . kv)/sqrt(hd), masked; invalid flags ----
    for (int s = tid; s < 512; s += 256) {
        const int r = s >> 7, h = (s >> 5) & 3, k = s & 31;
        const int m = m0 + r, bb = m >> 6, p = m & 63;
        const float* qt = sU + r * 512 + h * 128;
        const _Float16* kv = sKV + (r * 32 + k) * E2;
        float acc = 0.f;
#pragma unroll 8
        for (int j = 0; j < E2; ++j) acc += qt[j] * (float)kv[j];
        acc *= 0.1767766952966369f;               // 1/sqrt(32)
        if (mask[bb * PL + p * NNB + k] != 0) acc = -1e9f;
        sU2[s] = acc;                              // layout r*128 + h*32 + k == s
    }
    if (tid < 4) {
        const int m = m0 + tid, bb = m >> 6, p = m & 63;
        int inv = 1;
        for (int k = 0; k < NNB; ++k)
            if (mask[bb * PL + p * NNB + k] == 0) { inv = 0; break; }
        sInv[tid] = inv;
    }
    __syncthreads();

    // ---- Phase 6: softmax over 32, one (row,head) per thread ----
    if (tid < 16) {
        float* row = sU2 + tid * 32;
        float mx = row[0];
        for (int k = 1; k < NNB; ++k) mx = fmaxf(mx, row[k]);
        float sm = 0.f;
        for (int k = 0; k < NNB; ++k) { float e = __expf(row[k] - mx); row[k] = e; sm += e; }
        const float inv = 1.f / sm;
        for (int k = 0; k < NNB; ++k) row[k] *= inv;
    }
    __syncthreads();

    // ---- Phase 7: z[r][h][:] = sum_k attn * kv ----
    for (int s = tid; s < 2048; s += 256) {
        const int r = s >> 9, h = (s >> 7) & 3, jj = s & 127;
        const float* at = sU2 + r * 128 + h * 32;
        float acc = 0.f;
#pragma unroll 8
        for (int k = 0; k < NNB; ++k) acc += at[k] * (float)sKV[(r * 32 + k) * E2 + jj];
        sU[2048 + s] = acc;
    }
    __syncthreads();

    // ---- Phase 8: ctx = Wv_h z + bv ----
    for (int s = tid; s < 512; s += 256) {
        const int r = s >> 7, o = s & 127, h = o >> 5;
        const float* z = sU + 2048 + r * 512 + h * 128;
        const float* w = Wv + o * E2;
        float acc = bv[o];
#pragma unroll 8
        for (int j = 0; j < E2; ++j) acc += z[j] * w[j];
        sQ[s] = acc;                               // reuse sQ as ctx
    }
    __syncthreads();

    // ---- Phase 9: attn_out = Wo ctx + bo; zero invalid; store ----
    for (int s = tid; s < 512; s += 256) {
        const int r = s >> 7, o2 = s & 127;
        const float* w = Wo + o2 * E2;
        const float* cx = sQ + r * E2;
        float acc = bo[o2];
#pragma unroll 8
        for (int o = 0; o < E2; ++o) acc += w[o] * cx[o];
        if (sInv[r]) acc = 0.f;
        attnOut[(size_t)(m0 + r) * E2 + o2] = acc;
    }
}

// =====================================================================
// Kernel 2: final MLP  h = [embed_self, attn_out] @ W3a^T + b3a;
//           LN; tanh; @ W3b^T + b3b; tanh.  32 rows / block, WMMA GEMMs.
// =====================================================================
__global__ void __launch_bounds__(256)
hgcn_mlp_kernel(const float* __restrict__ memory,
                const int* __restrict__ nodes,
                const float* __restrict__ attnOut,
                const float* __restrict__ W3a, const float* __restrict__ b3a,
                const float* __restrict__ ln_g, const float* __restrict__ ln_b,
                const float* __restrict__ W3b, const float* __restrict__ b3b,
                float* __restrict__ out) {
    __shared__ _Float16 sIn[32 * 192];     // input tile (12KB); reused as tanh(LN(h)) f16 [32*96]
    __shared__ _Float16 sWf[36 * 32 * 16]; // weight fragments (36KB), W3a then W3b
    __shared__ float    sH[32 * 96];       // h f32 (12KB)

    const int tid = threadIdx.x;
    const int lane = tid & 31;
    const int wave = tid >> 5;
    const int m0 = blockIdx.x * 32;

    // stage W3a fragments: K=192 -> 6 ktiles, N=96 -> 6 ntiles
    stage_w_frag(sWf, W3a, 192, 6, 6 * 6 * 32, tid);
    // build input tile [32 x 192] f16
    for (int idx = tid; idx < 32 * 192; idx += 256) {
        const int r = idx / 192, c = idx - r * 192;
        const int m = m0 + r, bb = m >> 6, p = m & 63;
        float v;
        if (c < 64) v = memory[(size_t)nodes[bb * PP + p] * DD + c];
        else        v = attnOut[(size_t)m * E2 + (c - 64)];
        sIn[idx] = (_Float16)v;
    }
    __syncthreads();

    // GEMM1: [32,192] x [192->96] : 2 row-tiles x 6 col-tiles
    for (int t = wave; t < 12; t += 8) {
        const int rt = t / 6, nt = t - rt * 6;
        f32x8 c = zero8();
#pragma unroll
        for (int kt = 0; kt < 6; ++kt) {
            f16x16 a = load_a_frag(sIn, 192, rt * 16, kt * 32, lane);
            f16x16 b = load_b_frag(sWf + ((kt * 6 + nt) * 32 + lane) * 16);
            c = __builtin_amdgcn_wmma_f32_16x16x32_f16(false, a, false, b,
                                                       (short)0, c, false, false);
        }
        const int col = nt * 16 + (lane & 15);
        const float bv = b3a[col];
#pragma unroll
        for (int r = 0; r < 8; ++r) {
            const int row = rt * 16 + (lane >> 4) * 8 + r;
            sH[row * H1 + col] = c[r] + bv;
        }
    }
    __syncthreads();

    // LayerNorm + tanh (one thread per row), write f16 into sIn (stride 96)
    if (tid < 32) {
        const float* h = sH + tid * H1;
        float mu = 0.f;
        for (int j = 0; j < H1; ++j) mu += h[j];
        mu *= (1.f / H1);
        float var = 0.f;
        for (int j = 0; j < H1; ++j) { const float d = h[j] - mu; var += d * d; }
        var *= (1.f / H1);
        const float inv = rsqrtf(var + 1e-5f);
        _Float16* o = sIn + tid * H1;
        for (int j = 0; j < H1; ++j)
            o[j] = (_Float16)fast_tanh((h[j] - mu) * inv * ln_g[j] + ln_b[j]);
    }
    // stage W3b fragments (overwrites sWf; GEMM1 readers done at prior barrier):
    // K=96 -> 3 ktiles, N=64 -> 4 ntiles
    stage_w_frag(sWf, W3b, H1, 4, 3 * 4 * 32, tid);
    __syncthreads();

    // GEMM2: [32,96] x [96->64] : 2 row-tiles x 4 col-tiles -> one per wave
    for (int t = wave; t < 8; t += 8) {
        const int rt = t >> 2, nt = t & 3;
        f32x8 c = zero8();
#pragma unroll
        for (int kt = 0; kt < 3; ++kt) {
            f16x16 a = load_a_frag(sIn, H1, rt * 16, kt * 32, lane);
            f16x16 b = load_b_frag(sWf + ((kt * 4 + nt) * 32 + lane) * 16);
            c = __builtin_amdgcn_wmma_f32_16x16x32_f16(false, a, false, b,
                                                       (short)0, c, false, false);
        }
        const int col = nt * 16 + (lane & 15);
        const float bv = b3b[col];
#pragma unroll
        for (int r = 0; r < 8; ++r) {
            const int row = rt * 16 + (lane >> 4) * 8 + r;
            out[(size_t)(m0 + row) * DD + col] = fast_tanh(c[r] + bv);
        }
    }
}

extern "C" void kernel_launch(void* const* d_in, const int* in_sizes, int n_in,
                              void* d_out, int out_size, void* d_ws, size_t ws_size,
                              hipStream_t stream) {
    const float* memory     = (const float*)d_in[0];
    const float* times      = (const float*)d_in[1];
    const float* basis_freq = (const float*)d_in[2];
    const float* phase      = (const float*)d_in[3];
    const float* Wh  = (const float*)d_in[4];
    const float* bh  = (const float*)d_in[5];
    const float* Wq  = (const float*)d_in[6];
    const float* bq  = (const float*)d_in[7];
    const float* Wk  = (const float*)d_in[8];
    // bk (d_in[9]) is provably irrelevant: constant shift inside softmax rows.
    const float* Wv  = (const float*)d_in[10];
    const float* bv  = (const float*)d_in[11];
    const float* Wo  = (const float*)d_in[12];
    const float* bo  = (const float*)d_in[13];
    const float* W3a = (const float*)d_in[14];
    const float* b3a = (const float*)d_in[15];
    const float* ln_g = (const float*)d_in[16];
    const float* ln_b = (const float*)d_in[17];
    const float* W3b = (const float*)d_in[18];
    const float* b3b = (const float*)d_in[19];
    const int* nodes   = (const int*)d_in[20];
    const int* h_index = (const int*)d_in[21];
    const int* mask    = (const int*)d_in[22];

    float* attnOut = (float*)d_ws;   // [MROWS,128] f32 = 8 MB

    hgcn_attn_kernel<<<MROWS / 4, 256, 0, stream>>>(
        memory, times, basis_freq, phase, Wh, bh, Wq, bq, Wk, Wv, bv, Wo, bo,
        nodes, h_index, mask, attnOut);

    hgcn_mlp_kernel<<<MROWS / 32, 256, 0, stream>>>(
        memory, nodes, attnOut, W3a, b3a, ln_g, ln_b, W3b, b3b, (float*)d_out);
}
